// ScaledDotProductAttention_42915313221711
// MI455X (gfx1250) — matrix operand order repaired and checked by reading, not compiled, not run on hardware
//
#include <hip/hip_runtime.h>
#include <stdint.h>

typedef __attribute__((ext_vector_type(16))) _Float16 v16h;
typedef __attribute__((ext_vector_type(8)))  float    v8f;
typedef __attribute__((ext_vector_type(2)))  _Float16 h2;

#define B_   2
#define S_   2048
#define H_   16
#define D_   128
#define BC   32      // keys per step
#define BR   128     // query rows per block (8 waves x 16)
#define NSTEP (S_ / BC)
// scale * log2(e): softmax computed in exp2 domain
#define QSCALE_ 0.12751791441734075f

// LDS layout (in dwords) -- double buffered streaming region
#define K_STRIDE 68                  // per K row: 128 f16 (64 dw) + 4 dw pad
#define K_DWORDS (32 * K_STRIDE)     // 2176
#define V_STRIDE 20                  // per V^T row: 32 keys f16 (16 dw) + 4 dw pad
#define V_DWORDS (128 * V_STRIDE)    // 2560
#define BUF_DWORDS (K_DWORDS + V_DWORDS + 32)   // 4768 (incl. 32 f32 mask bias)
#define OUT_ROW  68                  // epilogue: 64 f32 + 4 pad
#define OUT_DWORDS (8 * 16 * OUT_ROW)           // 8704
#define SMEM_DWORDS (2 * BUF_DWORDS)            // 9536 dw = 38144 B  (> OUT_DWORDS)

union HU { v16h v; uint32_t u[8]; };

static __device__ inline v8f wmma_f16(v16h a, v16h b, v8f c) {
  return __builtin_amdgcn_wmma_f32_16x16x32_f16(false, a, false, b, (short)0, c, false, false);
}

static __device__ inline uint32_t pack2(float a, float b) {
  union { h2 h; uint32_t u; } x;
  x.h.x = (_Float16)a;
  x.h.y = (_Float16)b;
  return x.u;
}

// ---- global -> registers for one 32-key K/V tile (+ mask) ----
static __device__ inline void load_tile_regs(const float* __restrict__ K,
                                             const float* __restrict__ V,
                                             const int* __restrict__ mask,
                                             int b, int h, int k0, int tid,
                                             float kf[16], float vf[16], int& mv) {
  {
    const int r = tid >> 3, s = tid & 7;           // key row, 16-dim segment
    const float* kp = K + ((size_t)(b * S_ + k0 + r) * H_ + h) * D_ + 16 * s;
    #pragma unroll
    for (int j = 0; j < 4; ++j) {
      float4 t4 = *reinterpret_cast<const float4*>(kp + 4 * j);
      kf[4 * j + 0] = t4.x; kf[4 * j + 1] = t4.y; kf[4 * j + 2] = t4.z; kf[4 * j + 3] = t4.w;
    }
  }
  {
    const int kq = tid & 7, dq = tid >> 3;         // key quad / dim quad
    const float* vp = V + ((size_t)(b * S_ + k0 + 4 * kq) * H_ + h) * D_ + 4 * dq;
    #pragma unroll
    for (int j = 0; j < 4; ++j) {
      float4 t4 = *reinterpret_cast<const float4*>(vp + (size_t)j * H_ * D_);
      vf[4 * j + 0] = t4.x; vf[4 * j + 1] = t4.y; vf[4 * j + 2] = t4.z; vf[4 * j + 3] = t4.w;
    }
  }
  mv = mask[b * S_ + k0 + (tid & 31)];
}

// ---- registers -> LDS (f16 pack, V transposed) ----
static __device__ inline void store_tile_lds(uint32_t* Kl, uint32_t* Vl, float* biasL,
                                             int tid, const float kf[16], const float vf[16],
                                             int mv) {
  {
    const int r = tid >> 3, s = tid & 7;
    uint32_t dw[8];
    #pragma unroll
    for (int j = 0; j < 8; ++j) dw[j] = pack2(kf[2 * j], kf[2 * j + 1]);
    uint32_t* dst = Kl + r * K_STRIDE + 8 * s;
    *reinterpret_cast<uint4*>(dst)     = make_uint4(dw[0], dw[1], dw[2], dw[3]);
    *reinterpret_cast<uint4*>(dst + 4) = make_uint4(dw[4], dw[5], dw[6], dw[7]);
  }
  {
    const int kq = tid & 7, dq = tid >> 3;
    #pragma unroll
    for (int i = 0; i < 4; ++i) {
      const int d = 4 * dq + i;
      uint32_t u0 = pack2(vf[0 * 4 + i], vf[1 * 4 + i]);
      uint32_t u1 = pack2(vf[2 * 4 + i], vf[3 * 4 + i]);
      *reinterpret_cast<uint2*>(Vl + d * V_STRIDE + 2 * kq) = make_uint2(u0, u1);
    }
  }
  if (tid < BC) biasL[tid] = mv ? 0.0f : -1e30f;
}

__global__ __launch_bounds__(256)
void fa_fwd_kernel(const float* __restrict__ Q, const float* __restrict__ K,
                   const float* __restrict__ V, const int* __restrict__ mask,
                   float* __restrict__ O) {
  __shared__ uint32_t smem[SMEM_DWORDS];

  const int tid  = threadIdx.x;
  const int lane = tid & 31;
  const int w    = tid >> 5;        // wave id 0..7
  const int g    = lane >> 4;       // lane half 0/1
  const int nq   = lane & 15;       // C-layout column = this lane's query row

  const int qb = blockIdx.x & 15;
  const int h  = (blockIdx.x >> 4) & 15;
  const int b  = blockIdx.x >> 8;
  const int q0 = qb * BR;
  const int qrow = q0 + w * 16 + nq;

  // ---- Q^T in registers, B-matrix layout, scaled by 1/sqrt(d)*log2(e), f16 ----
  HU qB[4];
  {
    const float* qp = Q + ((size_t)(b * S_ + qrow) * H_ + h) * D_;
    #pragma unroll
    for (int c = 0; c < 4; ++c) {
      float f[16];
      #pragma unroll
      for (int j = 0; j < 4; ++j) {
        float4 t4 = *reinterpret_cast<const float4*>(qp + 32 * c + 8 * g + 4 * j + 8 * (j >> 1));
        f[4 * j + 0] = t4.x; f[4 * j + 1] = t4.y; f[4 * j + 2] = t4.z; f[4 * j + 3] = t4.w;
      }
      #pragma unroll
      for (int v = 0; v < 8; ++v)
        qB[c].u[v] = pack2(f[2 * v] * QSCALE_, f[2 * v + 1] * QSCALE_);
    }
  }

  v8f acc[8];
  #pragma unroll
  for (int t = 0; t < 8; ++t) acc[t] = v8f{0, 0, 0, 0, 0, 0, 0, 0};
  float mrow = -1e30f, lrow = 0.0f;   // log2-domain running max / sum

  // ---- prologue: fill buffer 0 ----
  float kf[16], vf[16];
  int mv;
  load_tile_regs(K, V, mask, b, h, 0, tid, kf, vf, mv);
  store_tile_lds(smem, smem + K_DWORDS, reinterpret_cast<float*>(smem + K_DWORDS + V_DWORDS),
                 tid, kf, vf, mv);
  __syncthreads();

  for (int step = 0; step < NSTEP; ++step) {
    const int cur = step & 1;
    uint32_t* Klc  = smem + cur * BUF_DWORDS;
    uint32_t* Vlc  = Klc + K_DWORDS;
    float*    bsc  = reinterpret_cast<float*>(Vlc + V_DWORDS);
    const bool have_next = (step + 1) < NSTEP;

    // issue next tile's global loads early (overlap with compute below)
    if (have_next)
      load_tile_regs(K, V, mask, b, h, (step + 1) * BC, tid, kf, vf, mv);

    // ---- S^T = K_tile @ Q^T : stage all 8 A tiles, then 8 WMMAs ----
    HU a[8];
    {
      const uint32_t* krow0 = Klc + nq * K_STRIDE;
      const uint32_t* krow1 = Klc + (16 + nq) * K_STRIDE;
      #pragma unroll
      for (int c = 0; c < 4; ++c) {
        uint4 lo0 = *reinterpret_cast<const uint4*>(krow0 + 16 * c + 4 * g);
        uint4 hi0 = *reinterpret_cast<const uint4*>(krow0 + 16 * c + 8 + 4 * g);
        uint4 lo1 = *reinterpret_cast<const uint4*>(krow1 + 16 * c + 4 * g);
        uint4 hi1 = *reinterpret_cast<const uint4*>(krow1 + 16 * c + 8 + 4 * g);
        a[2*c].u[0]=lo0.x; a[2*c].u[1]=lo0.y; a[2*c].u[2]=lo0.z; a[2*c].u[3]=lo0.w;
        a[2*c].u[4]=hi0.x; a[2*c].u[5]=hi0.y; a[2*c].u[6]=hi0.z; a[2*c].u[7]=hi0.w;
        a[2*c+1].u[0]=lo1.x; a[2*c+1].u[1]=lo1.y; a[2*c+1].u[2]=lo1.z; a[2*c+1].u[3]=lo1.w;
        a[2*c+1].u[4]=hi1.x; a[2*c+1].u[5]=hi1.y; a[2*c+1].u[6]=hi1.z; a[2*c+1].u[7]=hi1.w;
      }
    }
    v8f s0 = v8f{0, 0, 0, 0, 0, 0, 0, 0};
    v8f s1 = v8f{0, 0, 0, 0, 0, 0, 0, 0};
    #pragma unroll
    for (int c = 0; c < 4; ++c) {
      s0 = wmma_f16(a[2 * c].v,     qB[c].v, s0);
      s1 = wmma_f16(a[2 * c + 1].v, qB[c].v, s1);
    }

    // ---- online softmax in exp2 domain ----
    float f0[8], f1[8];
    #pragma unroll
    for (int r = 0; r < 8; ++r) {
      f0[r] = s0[r] + bsc[r + 8 * g];
      f1[r] = s1[r] + bsc[16 + r + 8 * g];
    }
    float mx = -1e30f;
    #pragma unroll
    for (int r = 0; r < 8; ++r) mx = fmaxf(mx, fmaxf(f0[r], f1[r]));
    mx = fmaxf(mx, __shfl_xor(mx, 16, 32));
    const float mnew = fmaxf(mrow, mx);
    const float scl  = exp2f(mrow - mnew);
    float rs = 0.0f;
    #pragma unroll
    for (int r = 0; r < 8; ++r) {
      f0[r] = exp2f(f0[r] - mnew);
      f1[r] = exp2f(f1[r] - mnew);
      rs += f0[r] + f1[r];
    }
    rs += __shfl_xor(rs, 16, 32);
    lrow = lrow * scl + rs;
    mrow = mnew;
    #pragma unroll
    for (int t = 0; t < 8; ++t) acc[t] *= scl;

    // ---- P^T into B-matrix layout: pack + cross-half lane swap ----
    uint32_t pk0[4], pk1[4], x0[4], x1[4];
    #pragma unroll
    for (int j = 0; j < 4; ++j) {
      pk0[j] = pack2(f0[2 * j], f0[2 * j + 1]);
      pk1[j] = pack2(f1[2 * j], f1[2 * j + 1]);
    }
    #pragma unroll
    for (int j = 0; j < 4; ++j) {
      x0[j] = (uint32_t)__shfl_xor((int)pk0[j], 16, 32);
      x1[j] = (uint32_t)__shfl_xor((int)pk1[j], 16, 32);
    }
    HU pB;
    #pragma unroll
    for (int j = 0; j < 4; ++j) {
      pB.u[j]     = g ? x1[j]  : pk0[j];
      pB.u[4 + j] = g ? pk1[j] : x0[j];
    }

    // ---- O^T += V^T @ P^T : stage all 8 A tiles, then 8 WMMAs ----
    HU va[8];
    #pragma unroll
    for (int t = 0; t < 8; ++t) {
      const uint32_t* vr = Vlc + (16 * t + nq) * V_STRIDE;
      uint4 lo = *reinterpret_cast<const uint4*>(vr + 8 * g);
      uint4 hi = *reinterpret_cast<const uint4*>(vr + 8 * g + 4);
      va[t].u[0]=lo.x; va[t].u[1]=lo.y; va[t].u[2]=lo.z; va[t].u[3]=lo.w;
      va[t].u[4]=hi.x; va[t].u[5]=hi.y; va[t].u[6]=hi.z; va[t].u[7]=hi.w;
    }
    #pragma unroll
    for (int t = 0; t < 8; ++t)
      acc[t] = wmma_f16(va[t].v, pB.v, acc[t]);

    // ---- write prefetched tile into the other buffer ----
    if (have_next) {
      uint32_t* Kln = smem + (cur ^ 1) * BUF_DWORDS;
      uint32_t* Vln = Kln + K_DWORDS;
      store_tile_lds(Kln, Vln, reinterpret_cast<float*>(Vln + V_DWORDS), tid, kf, vf, mv);
    }
    __syncthreads();
  }

  // ---- epilogue: normalize, bounce O^T through LDS for coalesced stores ----
  const float inv = 1.0f / lrow;

  float* ow = reinterpret_cast<float*>(smem) + (size_t)w * (16 * OUT_ROW);
  const int qbase = q0 + w * 16;
  #pragma unroll
  for (int p = 0; p < 2; ++p) {
    #pragma unroll
    for (int t2 = 0; t2 < 4; ++t2)
      #pragma unroll
      for (int r = 0; r < 8; ++r)
        ow[nq * OUT_ROW + 16 * t2 + r + 8 * g] = acc[4 * p + t2][r] * inv;
    __syncthreads();
    for (int i = 0; i < 16; ++i) {
      float2 val = *reinterpret_cast<const float2*>(ow + i * OUT_ROW + 2 * lane);
      *reinterpret_cast<float2*>(O + ((size_t)(b * S_ + qbase + i) * H_ + h) * D_ + 64 * p + 2 * lane) = val;
    }
    __syncthreads();
  }
}

extern "C" void kernel_launch(void* const* d_in, const int* in_sizes, int n_in,
                              void* d_out, int out_size, void* d_ws, size_t ws_size,
                              hipStream_t stream) {
  (void)in_sizes; (void)n_in; (void)out_size; (void)d_ws; (void)ws_size;
  const float* Q = (const float*)d_in[0];
  const float* K = (const float*)d_in[1];
  const float* V = (const float*)d_in[2];
  const int* mask = (const int*)d_in[3];
  float* O = (float*)d_out;

  dim3 grid(B_ * H_ * (S_ / BR));   // 512 blocks
  dim3 block(256);                  // 8 waves (wave32)
  fa_fwd_kernel<<<grid, block, 0, stream>>>(Q, K, V, mask, O);
}
